// Net_51445118271652
// MI455X (gfx1250) — matrix operand order repaired and checked by reading, not compiled, not run on hardware
//
#include <hip/hip_runtime.h>
#include <math.h>

// ---------------------------------------------------------------------------
// Graph-matching GNN forward for MI455X (gfx1250, wave32, WMMA).
// Every GEMM runs as  C = A[M,K] x B[N,K]^T  through V_WMMA_F32_16X16X32_BF16.
// Per lane: A fragment = two contiguous b128 loads, B fragment = one 32B load
// (2x b128).  Each wave owns a 32x32 C tile (2x2 fragments): 8 b128 loads ->
// 4 WMMAs per k-step, software-pipelined (next tile's loads issued before the
// current WMMAs).  Wave id is readfirstlane'd so all tile guards are scalar
// branches (EXEC stays all-ones for WMMA, no per-iteration exec juggling).
// The 4000x4000 operands (An, s) are L2-resident (192MB L2) -> compute-bound.
// ---------------------------------------------------------------------------

#define NN    2000
#define NT    4000
#define DIN   128
#define HID   512
#define FEAT  256
#define NLAY  3
#define ALPHA 200.0f
#define BSEPS 1e-4f

typedef __attribute__((ext_vector_type(16))) __bf16 v16bf;
typedef __attribute__((ext_vector_type(8)))  __bf16 v8bf;
typedef __attribute__((ext_vector_type(8)))  float  v8f;

__device__ __forceinline__ __bf16 f2bf(float f) {
  unsigned u = __float_as_uint(f);
  unsigned r = u + 0x7FFFu + ((u >> 16) & 1u);           // round-to-nearest-even
  return __builtin_bit_cast(__bf16, (unsigned short)(r >> 16));
}

// A fragment (16x32 bf16 row-major [M,K]); p = &A[row*lda + half*16 + kb]:
// two 16B runs at K = half*16 and K = half*16+8 (contiguous 32B).
__device__ __forceinline__ v16bf load_a(const __bf16* p) {
  v8bf lo = *(const v8bf*)p;
  v8bf hi = *(const v8bf*)(p + 8);
  return __builtin_shufflevector(lo, hi, 0,1,2,3,4,5,6,7,8,9,10,11,12,13,14,15);
}
// B fragment (32x16 bf16) with B stored [N,K]; p = &B[col*ldb + half*16 + kb]:
// one contiguous 32B run.
__device__ __forceinline__ v16bf load_b(const __bf16* p) {
  return *(const v16bf*)p;
}
__device__ __forceinline__ v8f load_cfrag(const float* C, int mbase, int nbase,
                                          int ldc, int half, int l16) {
  v8f r;
  const int row0 = mbase + half * 8;
  const int col  = nbase + l16;
#pragma unroll
  for (int j = 0; j < 8; ++j) r[j] = C[(long)(row0 + j) * ldc + col];
  return r;
}
__device__ __forceinline__ void epilogue(v8f acc, float* C, const float* bias,
                                         int mbase, int nbase, int ldc,
                                         int half, int l16, int act) {
  const int row0 = mbase + half * 8;
  const int col  = nbase + l16;
#pragma unroll
  for (int j = 0; j < 8; ++j) {
    float v = acc[j];
    if (bias) v += bias[col];
    if (act == 1)      v = fmaxf(v, 0.0f);
    else if (act == 2) v = 1.0f / (1.0f + expf(-v));
    C[(long)(row0 + j) * ldc + col] = v;
  }
}
#define WMMA_BF16(a, b, c) \
  __builtin_amdgcn_wmma_f32_16x16x32_bf16(false, (a), false, (b), (short)0, (c), false, false)

// ---------------------------------------------------------------------------
// WMMA GEMM:  C[M,N](f32) = act( A[M,K] * B[N,K]^T + bias + (accum ? C : 0) )
// act: 0=none 1=relu 2=sigmoid.  M,N multiples of 16; K multiple of 32.
// ---------------------------------------------------------------------------
__global__ __launch_bounds__(128)
void k_gemm_wmma(const __bf16* __restrict__ A, const __bf16* __restrict__ B,
                 float* __restrict__ C, const float* __restrict__ bias,
                 int M, int N, int K, int lda, int ldb, int ldc,
                 int accum, int act)
{
  const int lane = threadIdx.x & 31;
  // force wave id scalar -> all tile guards become s_cmp/s_cbranch
  const int wave = __builtin_amdgcn_readfirstlane(threadIdx.x >> 5);
  const int m0 = (blockIdx.y * 2 + (wave >> 1)) * 32;
  const int n0 = (blockIdx.x * 2 + (wave & 1)) * 32;
  if (m0 >= M || n0 >= N) return;
  const int half = lane >> 4;
  const int l16  = lane & 15;
  const bool mv1 = (m0 + 16) < M;                 // 2nd m-fragment valid
  const bool nv1 = (n0 + 16) < N;                 // 2nd n-fragment valid

  v8f zacc;
#pragma unroll
  for (int j = 0; j < 8; ++j) zacc[j] = 0.0f;

  v8f acc00 = accum                 ? load_cfrag(C, m0,      n0,      ldc, half, l16) : zacc;
  v8f acc01 = (accum && nv1)        ? load_cfrag(C, m0,      n0 + 16, ldc, half, l16) : zacc;
  v8f acc10 = (accum && mv1)        ? load_cfrag(C, m0 + 16, n0,      ldc, half, l16) : zacc;
  v8f acc11 = (accum && mv1 && nv1) ? load_cfrag(C, m0 + 16, n0 + 16, ldc, half, l16) : zacc;

  const __bf16* pa0 = A + (long)(m0 + l16)      * lda + half * 16;
  const __bf16* pa1 = A + (long)(m0 + l16 + 16) * lda + half * 16;
  const __bf16* pb0 = B + (long)(n0 + l16)      * ldb + half * 16;
  const __bf16* pb1 = B + (long)(n0 + l16 + 16) * ldb + half * 16;

  if (mv1 && nv1) {
    // ---- fast path: full 32x32 tile, software-pipelined, no conditionals ---
    v16bf a0 = load_a(pa0), a1 = load_a(pa1);
    v16bf b0 = load_b(pb0), b1 = load_b(pb1);
    for (int kb = 32; kb < K; kb += 32) {
      pa0 += 32; pa1 += 32; pb0 += 32; pb1 += 32;
      __builtin_prefetch((const void*)(pa0 + 32), 0, 1);   // global_prefetch_b8
      __builtin_prefetch((const void*)(pb0 + 32), 0, 1);
      v16bf na0 = load_a(pa0);
      v16bf na1 = load_a(pa1);
      v16bf nb0 = load_b(pb0);
      v16bf nb1 = load_b(pb1);
      acc00 = WMMA_BF16(a0, b0, acc00);
      acc01 = WMMA_BF16(a0, b1, acc01);
      acc10 = WMMA_BF16(a1, b0, acc10);
      acc11 = WMMA_BF16(a1, b1, acc11);
      a0 = na0; a1 = na1; b0 = nb0; b1 = nb1;
    }
    acc00 = WMMA_BF16(a0, b0, acc00);
    acc01 = WMMA_BF16(a0, b1, acc01);
    acc10 = WMMA_BF16(a1, b0, acc10);
    acc11 = WMMA_BF16(a1, b1, acc11);
  } else {
    // ---- edge path (M or N not a multiple of 32): rare, simple loop -------
    v16bf zfrag;
#pragma unroll
    for (int j = 0; j < 16; ++j) zfrag[j] = __builtin_bit_cast(__bf16, (unsigned short)0);
    for (int kb = 0; kb < K; kb += 32) {
      v16bf a0 = load_a(pa0 + kb);
      v16bf a1 = mv1 ? load_a(pa1 + kb) : zfrag;
      v16bf b0 = load_b(pb0 + kb);
      v16bf b1 = nv1 ? load_b(pb1 + kb) : zfrag;
      acc00 = WMMA_BF16(a0, b0, acc00);
      acc01 = WMMA_BF16(a0, b1, acc01);
      acc10 = WMMA_BF16(a1, b0, acc10);
      acc11 = WMMA_BF16(a1, b1, acc11);
    }
  }

  epilogue(acc00, C, bias, m0, n0, ldc, half, l16, act);
  if (nv1)        epilogue(acc01, C, bias, m0,      n0 + 16, ldc, half, l16, act);
  if (mv1)        epilogue(acc10, C, bias, m0 + 16, n0,      ldc, half, l16, act);
  if (mv1 && nv1) epilogue(acc11, C, bias, m0 + 16, n0 + 16, ldc, half, l16, act);
}

// ----------------------------- elementwise ---------------------------------
__global__ void k_f32_to_bf16(const float* __restrict__ in, __bf16* __restrict__ out, long n) {
  for (long i = (long)blockIdx.x * blockDim.x + threadIdx.x; i < n;
       i += (long)gridDim.x * blockDim.x)
    out[i] = f2bf(in[i]);
}
__global__ void k_copy_f32(float* __restrict__ dst, const float* __restrict__ src, long n) {
  for (long i = (long)blockIdx.x * blockDim.x + threadIdx.x; i < n;
       i += (long)gridDim.x * blockDim.x)
    dst[i] = src[i];
}
__global__ void k_zero_f32(float* __restrict__ dst, long n) {
  for (long i = (long)blockIdx.x * blockDim.x + threadIdx.x; i < n;
       i += (long)gridDim.x * blockDim.x)
    dst[i] = 0.0f;
}
// Tiled transpose-convert: outT[C,R] = bf16(in[R,C]); optionally also the
// straight row-major bf16 copy (out).  32x32 LDS tile, 32x8 threads.
__global__ __launch_bounds__(256)
void k_cvt_T_bf16(const float* __restrict__ in, __bf16* __restrict__ outT,
                  __bf16* __restrict__ out, int R, int C) {
  __shared__ float tile[32][33];
  const int tx = threadIdx.x & 31, ty = threadIdx.x >> 5;
  const int r0 = blockIdx.y * 32, c0 = blockIdx.x * 32;
#pragma unroll
  for (int i = 0; i < 32; i += 8) {
    int r = r0 + ty + i, c = c0 + tx;
    float v = (r < R && c < C) ? in[(long)r * C + c] : 0.0f;
    tile[ty + i][tx] = v;
    if (out && r < R && c < C) out[(long)r * C + c] = f2bf(v);
  }
  __syncthreads();
#pragma unroll
  for (int i = 0; i < 32; i += 8) {
    int c = c0 + ty + i, r = r0 + tx;
    if (c < C && r < R) outT[(long)c * R + r] = f2bf(tile[tx][ty + i]);
  }
}
// S = (A + A^T) * 0.5 -> bf16 (symmetric, so [K,N] == [N,K])
__global__ void k_sym_to_bf16(const float* __restrict__ A, __bf16* __restrict__ out, int dim) {
  int idx = blockIdx.x * blockDim.x + threadIdx.x;
  if (idx >= dim * dim) return;
  int i = idx / dim, j = idx % dim;
  out[idx] = f2bf(0.5f * (A[idx] + A[(long)j * dim + i]));
}
template <bool ABS>
__global__ void k_colsum(const float* __restrict__ A, float* __restrict__ out,
                         int rows, int cols) {
  int c = blockIdx.x * blockDim.x + threadIdx.x;
  if (c >= cols) return;
  float s = 0.0f;
  for (int r = 0; r < rows; ++r) {
    float v = A[(long)r * cols + c];
    s += ABS ? fabsf(v) : v;
  }
  out[c] = s;
}
__global__ void k_colnorm_bf16(const float* __restrict__ A, const float* __restrict__ cs,
                               __bf16* __restrict__ out, long rows, long cols) {
  long total = rows * cols;
  for (long i = (long)blockIdx.x * blockDim.x + threadIdx.x; i < total;
       i += (long)gridDim.x * blockDim.x) {
    long c = i % cols;
    out[i] = f2bf(A[i] / fmaxf(cs[c], 1e-12f));
  }
}
__global__ void k_colscale(float* __restrict__ S, const float* __restrict__ cs,
                           long rows, long cols) {
  long total = rows * cols;
  for (long i = (long)blockIdx.x * blockDim.x + threadIdx.x; i < total;
       i += (long)gridDim.x * blockDim.x) {
    long c = i % cols;
    S[i] /= cs[c];
  }
}
__global__ void k_mask_eps(float* __restrict__ S, int n1, int n2, int ncols,
                           long total, float eps) {
  for (long i = (long)blockIdx.x * blockDim.x + threadIdx.x; i < total;
       i += (long)gridDim.x * blockDim.x) {
    long r = i / ncols, c = i % ncols;
    bool blk = ((r < n1) && (c < n2)) || ((r >= n1) && (c >= n2));
    S[i] = blk ? (S[i] + eps) : 0.0f;
  }
}

// ------------------------------ row kernels --------------------------------
__global__ __launch_bounds__(256)
void k_rowsumsq(const float* __restrict__ X, float* __restrict__ out, int cols) {
  __shared__ float red[256];
  const float* row = X + (long)blockIdx.x * cols;
  float s = 0.0f;
  for (int c = threadIdx.x; c < cols; c += 256) { float v = row[c]; s += v * v; }
  red[threadIdx.x] = s; __syncthreads();
  for (int off = 128; off > 0; off >>= 1) {
    if (threadIdx.x < off) red[threadIdx.x] += red[threadIdx.x + off];
    __syncthreads();
  }
  if (threadIdx.x == 0) out[blockIdx.x] = red[0];
}
// in-place L2 row norm (f32) + bf16 copy + optional transposed bf16 copy
__global__ __launch_bounds__(256)
void k_l2norm_rows(float* __restrict__ X, __bf16* __restrict__ out_bf,
                   __bf16* __restrict__ out_bfT, int cols, int rtot) {
  __shared__ float red[256];
  const long rowi = blockIdx.x;
  float* row = X + rowi * cols;
  float s = 0.0f;
  for (int c = threadIdx.x; c < cols; c += 256) { float v = row[c]; s += v * v; }
  red[threadIdx.x] = s; __syncthreads();
  for (int off = 128; off > 0; off >>= 1) {
    if (threadIdx.x < off) red[threadIdx.x] += red[threadIdx.x + off];
    __syncthreads();
  }
  float inv = 1.0f / fmaxf(sqrtf(red[0]), 1e-12f);
  for (int c = threadIdx.x; c < cols; c += 256) {
    float v = row[c] * inv;
    row[c] = v;
    __bf16 bv = f2bf(v);
    if (out_bf)  out_bf[rowi * cols + c] = bv;
    if (out_bfT) out_bfT[(long)c * rtot + rowi] = bv;
  }
}
__global__ __launch_bounds__(256)
void k_softmax_rows(float* __restrict__ S, int cols, float alpha) {
  __shared__ float red[256];
  float* row = S + (long)blockIdx.x * cols;
  int tid = threadIdx.x;
  float m = -3.4e38f;
  for (int c = tid; c < cols; c += 256) m = fmaxf(m, row[c]);
  red[tid] = m; __syncthreads();
  for (int off = 128; off > 0; off >>= 1) {
    if (tid < off) red[tid] = fmaxf(red[tid], red[tid + off]);
    __syncthreads();
  }
  float mr = red[0]; __syncthreads();
  float s = 0.0f;
  for (int c = tid; c < cols; c += 256) s += expf(alpha * (row[c] - mr));
  red[tid] = s; __syncthreads();
  for (int off = 128; off > 0; off >>= 1) {
    if (tid < off) red[tid] += red[tid + off];
    __syncthreads();
  }
  float inv = 1.0f / red[0];
  for (int c = tid; c < cols; c += 256) row[c] = expf(alpha * (row[c] - mr)) * inv;
}
__global__ __launch_bounds__(256)
void k_rownorm(float* __restrict__ S, int cols) {
  __shared__ float red[256];
  float* row = S + (long)blockIdx.x * cols;
  float s = 0.0f;
  for (int c = threadIdx.x; c < cols; c += 256) s += row[c];
  red[threadIdx.x] = s; __syncthreads();
  for (int off = 128; off > 0; off >>= 1) {
    if (threadIdx.x < off) red[threadIdx.x] += red[threadIdx.x + off];
    __syncthreads();
  }
  float inv = 1.0f / red[0];
  for (int c = threadIdx.x; c < cols; c += 256) row[c] *= inv;
}

// -------------------- pairwise similarity finalization ---------------------
__global__ __launch_bounds__(256)
void k_dist_max(float* __restrict__ D, const float* __restrict__ x2,
                const float* __restrict__ y2, unsigned* __restrict__ gmax,
                int rows, int cols) {
  __shared__ float red[256];
  long total = (long)rows * cols;
  float lm = 0.0f;
  for (long i = (long)blockIdx.x * blockDim.x + threadIdx.x; i < total;
       i += (long)gridDim.x * blockDim.x) {
    int r = (int)(i / cols), c = (int)(i % cols);
    float d2 = x2[r] + y2[c] - 2.0f * D[i];
    float d = sqrtf(fmaxf(d2, 0.0f));
    D[i] = d;
    lm = fmaxf(lm, d);
  }
  red[threadIdx.x] = lm; __syncthreads();
  for (int off = 128; off > 0; off >>= 1) {
    if (threadIdx.x < off) red[threadIdx.x] = fmaxf(red[threadIdx.x], red[threadIdx.x + off]);
    __syncthreads();
  }
  if (threadIdx.x == 0) atomicMax(gmax, __float_as_uint(red[0]));
}
__global__ void k_sim_final(float* __restrict__ D, const unsigned* __restrict__ gmax, long n) {
  float m = fmaxf(__uint_as_float(*gmax), 1e-20f);
  for (long i = (long)blockIdx.x * blockDim.x + threadIdx.x; i < n;
       i += (long)gridDim.x * blockDim.x)
    D[i] = 1.0f - D[i] / m;
}

// ------------------------------- host side ---------------------------------
static inline unsigned gblk(long n) {
  long b = (n + 255) / 256;
  return (unsigned)(b > 32768 ? 32768 : b);
}
static void gemm(hipStream_t st, const __bf16* A, const __bf16* B, float* C,
                 const float* bias, int M, int N, int K, int lda, int ldb,
                 int ldc, int accum, int act) {
  dim3 grid((unsigned)((N + 63) / 64), (unsigned)((M + 63) / 64)), blk(128);
  k_gemm_wmma<<<grid, blk, 0, st>>>(A, B, C, bias, M, N, K, lda, ldb, ldc, accum, act);
}

extern "C" void kernel_launch(void* const* d_in, const int* in_sizes, int n_in,
                              void* d_out, int out_size, void* d_ws, size_t ws_size,
                              hipStream_t stream)
{
  (void)in_sizes; (void)n_in; (void)out_size; (void)ws_size;
  const float* emb1    = (const float*)d_in[0];
  const float* emb2    = (const float*)d_in[1];
  const float* eemb1   = (const float*)d_in[2];
  const float* eemb2   = (const float*)d_in[3];
  const float* A_src   = (const float*)d_in[4];
  const float* A_tgt   = (const float*)d_in[5];
  const float* fc1n_w  = (const float*)d_in[6];
  const float* fc1n_b  = (const float*)d_in[7];
  const float* fc2n_w  = (const float*)d_in[8];
  const float* fc2n_b  = (const float*)d_in[9];
  const float* fc1e_w  = (const float*)d_in[10];
  const float* fc1e_b  = (const float*)d_in[11];
  const float* fc2e_w  = (const float*)d_in[12];
  const float* fc2e_b  = (const float*)d_in[13];
  const float* gnn_a_w = (const float*)d_in[14];
  const float* gnn_a_b = (const float*)d_in[15];
  const float* gnn_u_w = (const float*)d_in[16];
  const float* gnn_u_b = (const float*)d_in[17];
  const float* aff_A   = (const float*)d_in[18];
  const float* cg_w    = (const float*)d_in[19];
  const float* cg_b    = (const float*)d_in[20];

  float* S  = (float*)d_out;                 // [NT,NT] final sinkhorn output
  float* Kp = S + (long)NT * NT;             // [NN,NN]
  float* Ke = Kp + (long)NN * NN;            // [NN,NN]

  // ---- workspace bump allocator ----
  char* wp = (char*)d_ws;
  auto alloc = [&](size_t bytes) -> void* {
    void* p = wp; wp += (bytes + 255) & ~(size_t)255; return p;
  };
  __bf16* An1_bf  = (__bf16*)alloc((size_t)NT * NT * 2);    // col-normalized adj
  __bf16* An2_bf  = (__bf16*)alloc((size_t)NT * NT * 2);
  __bf16* s_bf    = (__bf16*)alloc((size_t)NT * NT * 2);
  __bf16* sT_bf   = (__bf16*)alloc((size_t)NT * NT * 2);
  __bf16* x1_bf   = (__bf16*)alloc((size_t)NT * FEAT * 2);  // [NT,FEAT]
  __bf16* x2_bf   = (__bf16*)alloc((size_t)NT * FEAT * 2);
  __bf16* x1T_bf  = (__bf16*)alloc((size_t)NT * FEAT * 2);  // [FEAT,NT]
  __bf16* x2T_bf  = (__bf16*)alloc((size_t)NT * FEAT * 2);
  __bf16* axT_bf  = (__bf16*)alloc((size_t)NT * FEAT * 2);  // [FEAT,NT]
  __bf16* r1_bf   = (__bf16*)alloc((size_t)NT * FEAT * 2);
  __bf16* r2_bf   = (__bf16*)alloc((size_t)NT * FEAT * 2);
  __bf16* t_bf    = (__bf16*)alloc((size_t)NN * FEAT * 2);
  __bf16* emb_bf  = (__bf16*)alloc((size_t)NN * DIN * 2);
  __bf16* h_bf    = (__bf16*)alloc((size_t)NN * HID * 2);
  __bf16* e1n_bf  = (__bf16*)alloc((size_t)NN * FEAT * 2);
  __bf16* e2n_bf  = (__bf16*)alloc((size_t)NN * FEAT * 2);
  __bf16* e1e_bf  = (__bf16*)alloc((size_t)NN * FEAT * 2);
  __bf16* e2e_bf  = (__bf16*)alloc((size_t)NN * FEAT * 2);
  __bf16* w1nT    = (__bf16*)alloc((size_t)DIN * HID * 2);  // [HID,DIN]
  __bf16* w2nT    = (__bf16*)alloc((size_t)HID * FEAT * 2); // [FEAT,HID]
  __bf16* w1eT    = (__bf16*)alloc((size_t)DIN * HID * 2);
  __bf16* w2eT    = (__bf16*)alloc((size_t)HID * FEAT * 2);
  __bf16* gaT     = (__bf16*)alloc((size_t)NLAY * FEAT * FEAT * 2);
  __bf16* guT     = (__bf16*)alloc((size_t)NLAY * FEAT * FEAT * 2);
  __bf16* cgT_top = (__bf16*)alloc((size_t)FEAT * FEAT * 2);
  __bf16* cgT_bot = (__bf16*)alloc((size_t)FEAT * FEAT * 2);
  __bf16* affS_bf = (__bf16*)alloc((size_t)FEAT * FEAT * 2); // symmetric
  float*  h_f     = (float*)alloc((size_t)NN * HID * 4);
  float*  e1n_f   = (float*)alloc((size_t)NN * FEAT * 4);
  float*  e2n_f   = (float*)alloc((size_t)NN * FEAT * 4);
  float*  e1e_f   = (float*)alloc((size_t)NN * FEAT * 4);
  float*  e2e_f   = (float*)alloc((size_t)NN * FEAT * 4);
  float*  xf_tmp  = (float*)alloc((size_t)NT * FEAT * 4);
  float*  y_f     = (float*)alloc((size_t)NT * FEAT * 4);
  float*  r_f     = (float*)alloc((size_t)NT * FEAT * 4);
  float*  t_f     = (float*)alloc((size_t)NN * FEAT * 4);
  float*  colsum  = (float*)alloc((size_t)NT * 4);
  float*  rsq1    = (float*)alloc((size_t)NN * 4);
  float*  rsq2    = (float*)alloc((size_t)NN * 4);
  unsigned* gmax  = (unsigned*)alloc(256);

  auto cvt = [&](const float* in, __bf16* out, long n) {
    k_f32_to_bf16<<<gblk(n), 256, 0, stream>>>(in, out, n);
  };
  auto tcvt = [&](const float* in, __bf16* outT, __bf16* out, int R, int C) {
    dim3 g((unsigned)((C + 31) / 32), (unsigned)((R + 31) / 32)), b(256);
    k_cvt_T_bf16<<<g, b, 0, stream>>>(in, outT, out, R, C);
  };

  // ---- weights -> transposed bf16 ([N,K] layouts) ----
  tcvt(fc1n_w, w1nT, nullptr, DIN, HID);
  tcvt(fc2n_w, w2nT, nullptr, HID, FEAT);
  tcvt(fc1e_w, w1eT, nullptr, DIN, HID);
  tcvt(fc2e_w, w2eT, nullptr, HID, FEAT);
  for (int i = 0; i < NLAY; ++i) {
    tcvt(gnn_a_w + (long)i * FEAT * FEAT, gaT + (long)i * FEAT * FEAT, nullptr, FEAT, FEAT);
    tcvt(gnn_u_w + (long)i * FEAT * FEAT, guT + (long)i * FEAT * FEAT, nullptr, FEAT, FEAT);
  }
  tcvt(cg_w,                     cgT_top, nullptr, FEAT, FEAT);
  tcvt(cg_w + (long)FEAT * FEAT, cgT_bot, nullptr, FEAT, FEAT);

  // ---- node / edge embedding MLPs ----
  struct Path { const float* in; const __bf16* w1; const float* b1;
                const __bf16* w2; const float* b2; float* out_f; __bf16* out_bf; int act; };
  Path paths[4] = {
    { emb1,  w1nT, fc1n_b, w2nT, fc2n_b, e1n_f, e1n_bf, 0 },  // l2norm after
    { emb2,  w1nT, fc1n_b, w2nT, fc2n_b, e2n_f, e2n_bf, 0 },
    { eemb1, w1eT, fc1e_b, w2eT, fc2e_b, e1e_f, e1e_bf, 2 },  // sigmoid fused
    { eemb2, w1eT, fc1e_b, w2eT, fc2e_b, e2e_f, e2e_bf, 2 },
  };
  for (int p = 0; p < 4; ++p) {
    cvt(paths[p].in, emb_bf, (long)NN * DIN);
    gemm(stream, emb_bf, paths[p].w1, h_f, paths[p].b1,
         NN, HID, DIN, DIN, DIN, HID, 0, 1);                  // relu
    cvt(h_f, h_bf, (long)NN * HID);
    gemm(stream, h_bf, paths[p].w2, paths[p].out_f, paths[p].b2,
         NN, FEAT, HID, HID, HID, FEAT, 0, paths[p].act);
    if (paths[p].act == 0)
      k_l2norm_rows<<<NN, 256, 0, stream>>>(paths[p].out_f, paths[p].out_bf, nullptr, FEAT, NN);
    else
      cvt(paths[p].out_f, paths[p].out_bf, (long)NN * FEAT);
  }

  // ---- pairwise similarity Kp, Ke (1 - dist/max) ----
  struct Sim { const __bf16* xb; const __bf16* yb; const float* xf; const float* yf; float* out; };
  Sim sims[2] = { { e1n_bf, e2n_bf, e1n_f, e2n_f, Kp },
                  { e1e_bf, e2e_bf, e1e_f, e2e_f, Ke } };
  for (int q = 0; q < 2; ++q) {
    gemm(stream, sims[q].xb, sims[q].yb, sims[q].out, nullptr,
         NN, NN, FEAT, FEAT, FEAT, NN, 0, 0);                 // X*Y^T ([N,K] B)
    k_rowsumsq<<<NN, 256, 0, stream>>>(sims[q].xf, rsq1, FEAT);
    k_rowsumsq<<<NN, 256, 0, stream>>>(sims[q].yf, rsq2, FEAT);
    k_zero_f32<<<1, 32, 0, stream>>>((float*)gmax, 1);
    k_dist_max<<<gblk((long)NN * NN), 256, 0, stream>>>(sims[q].out, rsq1, rsq2, gmax, NN, NN);
    k_sim_final<<<gblk((long)NN * NN), 256, 0, stream>>>(sims[q].out, gmax, (long)NN * NN);
  }

  // ---- concat [node; edge] -> x1, x2 (bf16 + transposed bf16) ----
  k_copy_f32<<<gblk((long)NN * FEAT), 256, 0, stream>>>(xf_tmp, e1n_f, (long)NN * FEAT);
  k_copy_f32<<<gblk((long)NN * FEAT), 256, 0, stream>>>(xf_tmp + (long)NN * FEAT, e1e_f, (long)NN * FEAT);
  tcvt(xf_tmp, x1T_bf, x1_bf, NT, FEAT);
  k_copy_f32<<<gblk((long)NN * FEAT), 256, 0, stream>>>(xf_tmp, e2n_f, (long)NN * FEAT);
  k_copy_f32<<<gblk((long)NN * FEAT), 256, 0, stream>>>(xf_tmp + (long)NN * FEAT, e2e_f, (long)NN * FEAT);
  tcvt(xf_tmp, x2T_bf, x2_bf, NT, FEAT);

  // ---- column-L1-normalized adjacency -> bf16 (layer-invariant) ----
  k_colsum<true><<<(NT + 255) / 256, 256, 0, stream>>>(A_src, colsum, NT, NT);
  k_colnorm_bf16<<<gblk((long)NT * NT), 256, 0, stream>>>(A_src, colsum, An1_bf, NT, NT);
  k_colsum<true><<<(NT + 255) / 256, 256, 0, stream>>>(A_tgt, colsum, NT, NT);
  k_colnorm_bf16<<<gblk((long)NT * NT), 256, 0, stream>>>(A_tgt, colsum, An2_bf, NT, NT);

  // ---- GNN layers ----
  for (int i = 0; i < NLAY; ++i) {
    const __bf16* waT = gaT + (long)i * FEAT * FEAT;
    const __bf16* wuT = guT + (long)i * FEAT * FEAT;
    const float*  ba  = gnn_a_b + (long)i * FEAT;
    const float*  bu  = gnn_u_b + (long)i * FEAT;

    for (int side = 0; side < 2; ++side) {
      __bf16* xbf  = side ? x2_bf  : x1_bf;
      __bf16* xTbf = side ? x2T_bf : x1T_bf;
      const __bf16* An = side ? An2_bf : An1_bf;
      // ax = relu(x @ w_a + b_a)  -> transposed bf16 for B-side of An@ax
      gemm(stream, xbf, waT, r_f, ba, NT, FEAT, FEAT, FEAT, FEAT, FEAT, 0, 1);
      tcvt(r_f, axT_bf, nullptr, NT, FEAT);
      // y = relu(x @ w_u + b_u)
      gemm(stream, xbf, wuT, y_f, bu, NT, FEAT, FEAT, FEAT, FEAT, FEAT, 0, 1);
      // y += An @ ax   (4000x4000x256; An L2-resident)
      gemm(stream, An, axT_bf, y_f, nullptr, NT, FEAT, NT, NT, NT, FEAT, 1, 0);
      // x = l2norm(y)  (bf16 + bf16^T for downstream GEMMs)
      k_l2norm_rows<<<NT, 256, 0, stream>>>(y_f, xbf, xTbf, FEAT, NT);
    }

    // ---- affinity: block-diagonal s = X * sym(A_i) * Y^T ----
    k_sym_to_bf16<<<(FEAT * FEAT + 255) / 256, 256, 0, stream>>>(
        aff_A + (long)i * FEAT * FEAT, affS_bf, FEAT);
    k_zero_f32<<<gblk((long)NT * NT), 256, 0, stream>>>(S, (long)NT * NT);
    // upper (node) block
    gemm(stream, x1_bf, affS_bf, t_f, nullptr, NN, FEAT, FEAT, FEAT, FEAT, FEAT, 0, 0);
    cvt(t_f, t_bf, (long)NN * FEAT);
    gemm(stream, t_bf, x2_bf, S, nullptr, NN, NN, FEAT, FEAT, FEAT, NT, 0, 0);
    // lower (edge) block
    gemm(stream, x1_bf + (long)NN * FEAT, affS_bf, t_f, nullptr, NN, FEAT, FEAT, FEAT, FEAT, FEAT, 0, 0);
    cvt(t_f, t_bf, (long)NN * FEAT);
    gemm(stream, t_bf, x2_bf + (long)NN * FEAT, S + (long)NN * NT + NN, nullptr,
         NN, NN, FEAT, FEAT, FEAT, NT, 0, 0);

    // ---- Norm (softmax * ALPHA) + Sinkhorn ----
    k_softmax_rows<<<NT, 256, 0, stream>>>(S, NT, ALPHA);
    k_mask_eps<<<gblk((long)NT * NT), 256, 0, stream>>>(S, NN, NN, NT, (long)NT * NT, BSEPS);
    for (int it = 0; it < 10; ++it) {
      if ((it & 1) == 0) {
        k_colsum<false><<<(NT + 255) / 256, 256, 0, stream>>>(S, colsum, NT, NT);
        k_colscale<<<gblk((long)NT * NT), 256, 0, stream>>>(S, colsum, NT, NT);
      } else {
        k_rownorm<<<NT, 256, 0, stream>>>(S, NT);
      }
    }

    // ---- cross-graph update at i == NLAY-2 ----
    if (i == NLAY - 2) {
      tcvt(S, sT_bf, s_bf, NT, NT);        // s and s^T, both row-major bf16
      // r1 = s @ x2 ; r2 = s^T @ x1  (before overwriting x1/x2)
      gemm(stream, s_bf, x2T_bf, r_f, nullptr, NT, FEAT, NT, NT, NT, FEAT, 0, 0);
      cvt(r_f, r1_bf, (long)NT * FEAT);
      gemm(stream, sT_bf, x1T_bf, r_f, nullptr, NT, FEAT, NT, NT, NT, FEAT, 0, 0);
      cvt(r_f, r2_bf, (long)NT * FEAT);
      // x1' = x1@W_top + r1@W_bot + b
      gemm(stream, x1_bf, cgT_top, xf_tmp, cg_b, NT, FEAT, FEAT, FEAT, FEAT, FEAT, 0, 0);
      gemm(stream, r1_bf, cgT_bot, xf_tmp, nullptr, NT, FEAT, FEAT, FEAT, FEAT, FEAT, 1, 0);
      tcvt(xf_tmp, x1T_bf, x1_bf, NT, FEAT);
      // x2' = x2@W_top + r2@W_bot + b
      gemm(stream, x2_bf, cgT_top, xf_tmp, cg_b, NT, FEAT, FEAT, FEAT, FEAT, FEAT, 0, 0);
      gemm(stream, r2_bf, cgT_bot, xf_tmp, nullptr, NT, FEAT, FEAT, FEAT, FEAT, FEAT, 1, 0);
      tcvt(xf_tmp, x2T_bf, x2_bf, NT, FEAT);
    }
  }
  // final S (layer-3 sinkhorn output) already lives in d_out[0 : NT*NT].
}
